// PointNetAFP_58317065945849
// MI455X (gfx1250) — matrix operand order repaired and checked by reading, not compiled, not run on hardware
//
#include <hip/hip_runtime.h>
#include <hip/hip_bf16.h>
#include <math.h>

typedef __attribute__((ext_vector_type(16))) _Float16 v16h;
typedef __attribute__((ext_vector_type(8)))  _Float16 v8h;
typedef __attribute__((ext_vector_type(8)))  float    v8f;

#define BATCH 8
#define NPTS0 2048

__host__ __device__ static inline int rup32(int x) { return (x + 31) & ~31; }

// ---------------------------------------------------------------------------
// Weight prep: W (K x N, f32 row-major) -> Wt (Npad x Kpad, f16), transposed +
// zero padded so each WMMA B-fragment lane load is 16 contiguous halves.
// Bias zero-padded to Npad so padded output columns compute exactly 0.
// ---------------------------------------------------------------------------
__global__ void prep_weight_kernel(const float* __restrict__ W,
                                   const float* __restrict__ bsrc,
                                   int K, int N, int Kp, int Np,
                                   _Float16* __restrict__ Wt,
                                   float* __restrict__ Bp) {
  int t = blockIdx.x * blockDim.x + threadIdx.x;
  int total = Np * Kp;
  if (t < total) {
    int n = t / Kp, k = t % Kp;
    float v = (k < K && n < N) ? W[(size_t)k * N + n] : 0.f;
    Wt[t] = (_Float16)v;
  }
  if (t < Np) Bp[t] = (t < N) ? bsrc[t] : 0.f;
}

// ---------------------------------------------------------------------------
// Farthest point sampling, one block per batch. Matches reference semantics:
// record current farthest, min-update dists, argmax (first occurrence) next.
// ---------------------------------------------------------------------------
__global__ void __launch_bounds__(256) fps_kernel(const float* __restrict__ xyz,
                                                  int Npts, int npoint,
                                                  int* __restrict__ outIdx,
                                                  float* __restrict__ outXyz) {
  int b = blockIdx.x;
  int tid = threadIdx.x;
  __shared__ float dist[NPTS0];
  __shared__ float rv[256];
  __shared__ int   ri[256];
  __shared__ int   sIdx[512];
  __shared__ float cen[3];
  __shared__ int   cur;

  const float* bx = xyz + (size_t)b * Npts * 3;
  for (int p = tid; p < Npts; p += blockDim.x) dist[p] = 1e10f;
  if (tid == 0) cur = 0;
  __syncthreads();

  for (int it = 0; it < npoint; ++it) {
    if (tid == 0) {
      sIdx[it] = cur;
      cen[0] = bx[cur * 3 + 0];
      cen[1] = bx[cur * 3 + 1];
      cen[2] = bx[cur * 3 + 2];
    }
    __syncthreads();
    float bv = -1.f;
    int bi = 0;
    for (int p = tid; p < Npts; p += blockDim.x) {
      float dx = bx[p * 3 + 0] - cen[0];
      float dy = bx[p * 3 + 1] - cen[1];
      float dz = bx[p * 3 + 2] - cen[2];
      float d = dx * dx + dy * dy + dz * dz;
      float dm = dist[p];
      if (d < dm) { dm = d; dist[p] = d; }
      if (dm > bv) { bv = dm; bi = p; }   // ascending p: first-max kept
    }
    rv[tid] = bv; ri[tid] = bi;
    __syncthreads();
    for (int s = blockDim.x >> 1; s > 0; s >>= 1) {
      if (tid < s) {
        if (rv[tid + s] > rv[tid] ||
            (rv[tid + s] == rv[tid] && ri[tid + s] < ri[tid])) {
          rv[tid] = rv[tid + s]; ri[tid] = ri[tid + s];
        }
      }
      __syncthreads();
    }
    if (tid == 0) cur = ri[0];
    __syncthreads();
  }
  for (int i = tid; i < npoint; i += blockDim.x) {
    int p = sIdx[i];
    outIdx[(size_t)b * npoint + i] = p;
    outXyz[((size_t)b * npoint + i) * 3 + 0] = bx[p * 3 + 0];
    outXyz[((size_t)b * npoint + i) * 3 + 1] = bx[p * 3 + 1];
    outXyz[((size_t)b * npoint + i) * 3 + 2] = bx[p * 3 + 2];
  }
}

// ---------------------------------------------------------------------------
// Ball query, one thread per centroid. First `ns` in-radius indices in
// ascending order; pad with first index (or 0 if none) — reference semantics.
// ---------------------------------------------------------------------------
__global__ void ball_query_kernel(const float* __restrict__ xyz,
                                  const float* __restrict__ newXyz,
                                  int Npts, int M, float r2, int ns,
                                  int* __restrict__ idx) {
  int t = blockIdx.x * blockDim.x + threadIdx.x;
  if (t >= BATCH * M) return;
  int b = t / M, m = t % M;
  const float* bx = xyz + (size_t)b * Npts * 3;
  float c0 = newXyz[((size_t)b * M + m) * 3 + 0];
  float c1 = newXyz[((size_t)b * M + m) * 3 + 1];
  float c2 = newXyz[((size_t)b * M + m) * 3 + 2];
  int* out = idx + ((size_t)b * M + m) * ns;
  int cnt = 0, first = -1;
  for (int p = 0; p < Npts && cnt < ns; ++p) {
    float dx = bx[p * 3 + 0] - c0;
    float dy = bx[p * 3 + 1] - c1;
    float dz = bx[p * 3 + 2] - c2;
    if (dx * dx + dy * dy + dz * dz < r2) {
      if (first < 0) first = p;
      out[cnt++] = p;
    }
  }
  int pad = (first < 0) ? 0 : first;
  for (int j = cnt; j < ns; ++j) out[j] = pad;
}

// ---------------------------------------------------------------------------
// One MLP layer (relu(X @ W^T + b)) over `rows` rows held in LDS.
// - A fragments (contiguous k = 16*hf + s, matching B) via two ds_load_b128 per K-step,
//   loaded ONCE and reused for TWO column tiles (two accumulators -> two
//   independent WMMA chains, better co-execution / latency hiding).
// - B fragments are 32B contiguous loads of the pre-transposed padded
//   weights; W/Bb arrive straight from kernel args so the kernarg->global
//   promotion fires (global_load_b128 instead of flat_load_b128, keeping
//   the weight stream off DScnt).
// - Software-pipelined K loop: next K-step's fragments are issued before the
//   current WMMAs so the compiler can use partial s_wait counts.
// Tile loops are wave-uniform -> EXEC all ones around WMMA (ISA requirement).
// ---------------------------------------------------------------------------
__device__ __forceinline__ void mlp_layer(const _Float16* __restrict__ W,
                                          const float* __restrict__ Bb, int N,
                                          int kin, int rows,
                                          const _Float16* __restrict__ inB,
                                          int inW, _Float16* __restrict__ outB,
                                          int outW) {
  int tid = threadIdx.x;
  int lane = tid & 31;
  int wave = tid >> 5;
  int nwaves = blockDim.x >> 5;
  int hf = lane >> 4;
  int l16 = lane & 15;
  int Kp = rup32(kin);
  int Np = rup32(N);
  int rowTiles = rows >> 4;
  int colPairs = Np >> 5;   // two 16-wide column tiles per iteration

  for (int tile = wave; tile < rowTiles * colPairs; tile += nwaves) {
    int rt = tile % rowTiles;
    int cp = tile / rowTiles;
    int rowBase = rt << 4;
    int col0 = (cp << 5) + l16;
    int col1 = col0 + 16;

    v8f acc0, acc1;
    float bias0 = Bb[col0];
    float bias1 = Bb[col1];
#pragma unroll
    for (int i = 0; i < 8; ++i) { acc0[i] = bias0; acc1[i] = bias1; }

    const _Float16* arow = inB + (rowBase + l16) * inW + hf * 16;
    const _Float16* bc0 = W + (size_t)col0 * Kp + hf * 16;
    const _Float16* bc1 = W + (size_t)col1 * Kp + hf * 16;

    // prologue: K-step 0 fragments
    v8h lo = *(const v8h*)(arow);
    v8h hi = *(const v8h*)(arow + 8);
    v16h bf0 = *(const v16h*)(bc0);
    v16h bf1 = *(const v16h*)(bc1);

    for (int k0 = 32; k0 < Kp; k0 += 32) {
      v8h lo_n = *(const v8h*)(arow + k0);
      v8h hi_n = *(const v8h*)(arow + k0 + 8);
      v16h b0_n = *(const v16h*)(bc0 + k0);
      v16h b1_n = *(const v16h*)(bc1 + k0);
      v16h a;
#pragma unroll
      for (int i = 0; i < 8; ++i) { a[i] = lo[i]; a[i + 8] = hi[i]; }
      acc0 = __builtin_amdgcn_wmma_f32_16x16x32_f16(false, a, false, bf0,
                                                    (short)0, acc0, false,
                                                    false);
      acc1 = __builtin_amdgcn_wmma_f32_16x16x32_f16(false, a, false, bf1,
                                                    (short)0, acc1, false,
                                                    false);
      lo = lo_n; hi = hi_n; bf0 = b0_n; bf1 = b1_n;
    }
    {
      v16h a;
#pragma unroll
      for (int i = 0; i < 8; ++i) { a[i] = lo[i]; a[i + 8] = hi[i]; }
      acc0 = __builtin_amdgcn_wmma_f32_16x16x32_f16(false, a, false, bf0,
                                                    (short)0, acc0, false,
                                                    false);
      acc1 = __builtin_amdgcn_wmma_f32_16x16x32_f16(false, a, false, bf1,
                                                    (short)0, acc1, false,
                                                    false);
    }
    // C/D layout: lanes0-15 -> M=i, lanes16-31 -> M=i+8
#pragma unroll
    for (int i = 0; i < 8; ++i) {
      int rr = rowBase + hf * 8 + i;
      float v0 = acc0[i]; v0 = v0 > 0.f ? v0 : 0.f;
      float v1 = acc1[i]; v1 = v1 > 0.f ? v1 : 0.f;
      outB[rr * outW + col0] = (_Float16)v0;
      outB[rr * outW + col1] = (_Float16)v1;
    }
  }
}

// ---------------------------------------------------------------------------
// Fused set-abstraction kernel: one block per (batch, centroid).
// Gather grouped inputs into LDS (f16), run 3-4 MLP layers via WMMA
// ping-ponging X/Y LDS buffers, column-max pool, write pooled features.
// ---------------------------------------------------------------------------
template <int XW, int YW, int MAXROWS>
__global__ void __launch_bounds__(256) fused_sa_kernel(
    const float* __restrict__ xyz, const float* __restrict__ feats,
    const float* __restrict__ newXyz, const int* __restrict__ idxbuf,
    int Npts, int M, int nsample, int C, int featsFirst,
    const _Float16* __restrict__ w0, const float* __restrict__ b0, int n0,
    const _Float16* __restrict__ w1, const float* __restrict__ b1, int n1,
    const _Float16* __restrict__ w2, const float* __restrict__ b2, int n2,
    const _Float16* __restrict__ w3, const float* __restrict__ b3, int n3,
    float* __restrict__ outFeat, int Ctot, int coff) {
  __shared__ __align__(16) _Float16 Xs[MAXROWS * XW];
  __shared__ __align__(16) _Float16 Ys[MAXROWS * YW];
  __shared__ float colmax[256];

  int tid = threadIdx.x;
  int m = blockIdx.x % M;
  int b = blockIdx.x / M;

  int L = (n3 > 0) ? 4 : 3;
  int kin0 = C + 3;
  int Kp0 = rup32(kin0);
  int finalN = (n3 > 0) ? n3 : n2;

  colmax[tid] = 0.f;   // relu outputs >= 0, so 0-init matches max semantics
  __syncthreads();

  for (int cs = 0; cs < nsample; cs += MAXROWS) {
    int rows = nsample - cs;
    if (rows > MAXROWS) rows = MAXROWS;   // always a multiple of 16 here

    // ---- gather grouped inputs for this row chunk into Xs (f16) ----
    for (int t = tid; t < rows * Kp0; t += blockDim.x) {
      int r = t / Kp0, k = t % Kp0;
      int j = cs + r;
      int p = idxbuf ? idxbuf[((size_t)b * M + m) * nsample + j] : j;
      float v = 0.f;
      if (k < kin0) {
        if (featsFirst) {   // sa_msg: [feats, xyz - center]
          if (k < C) {
            v = feats[((size_t)b * Npts + p) * C + k];
          } else {
            int kk = k - C;
            float ce = newXyz ? newXyz[((size_t)b * M + m) * 3 + kk] : 0.f;
            v = xyz[((size_t)b * Npts + p) * 3 + kk] - ce;
          }
        } else {            // group_all: [xyz, feats]
          if (k < 3) {
            float ce = newXyz ? newXyz[((size_t)b * M + m) * 3 + k] : 0.f;
            v = xyz[((size_t)b * Npts + p) * 3 + k] - ce;
          } else {
            v = feats[((size_t)b * Npts + p) * C + (k - 3)];
          }
        }
      }
      Xs[r * XW + k] = (_Float16)v;
    }
    __syncthreads();

    // ---- MLP chain (unrolled so W pointers stay kernarg-direct) ----
    mlp_layer(w0, b0, n0, kin0, rows, Xs, XW, Ys, YW);
    __syncthreads();
    mlp_layer(w1, b1, n1, n0, rows, Ys, YW, Xs, XW);
    __syncthreads();
    mlp_layer(w2, b2, n2, n1, rows, Xs, XW, Ys, YW);
    __syncthreads();
    if (n3 > 0) {
      mlp_layer(w3, b3, n3, n2, rows, Ys, YW, Xs, XW);
      __syncthreads();
    }

    // ---- column max-pool update for this chunk ----
    const _Float16* fin = (L & 1) ? Ys : Xs;   // L=3 -> Ys, L=4 -> Xs
    int finW = (L & 1) ? YW : XW;
    for (int c = tid; c < finalN; c += blockDim.x) {
      float mx = colmax[c];
      for (int r = 0; r < rows; ++r) {
        float v = (float)fin[r * finW + c];
        if (v > mx) mx = v;
      }
      colmax[c] = mx;
    }
    __syncthreads();
  }

  for (int c = tid; c < finalN; c += blockDim.x)
    outFeat[((size_t)b * M + m) * Ctot + coff + c] = colmax[c];
}

// ---------------------------------------------------------------------------
// Row-wise L2 normalize of the (B x 128) embedding, in place.
// ---------------------------------------------------------------------------
__global__ void __launch_bounds__(128) l2norm_kernel(float* __restrict__ out,
                                                     int D) {
  __shared__ float red[128];
  int b = blockIdx.x;
  int tid = threadIdx.x;
  float v = out[(size_t)b * D + tid];
  red[tid] = v * v;
  __syncthreads();
  for (int s = 64; s > 0; s >>= 1) {
    if (tid < s) red[tid] += red[tid + s];
    __syncthreads();
  }
  float nrm = sqrtf(red[0]);
  float den = nrm > 1e-12f ? nrm : 1e-12f;
  out[(size_t)b * D + tid] = v / den;
}

// ---------------------------------------------------------------------------
extern "C" void kernel_launch(void* const* d_in, const int* in_sizes, int n_in,
                              void* d_out, int out_size, void* d_ws,
                              size_t ws_size, hipStream_t stream) {
  (void)in_sizes; (void)n_in; (void)out_size; (void)ws_size;
  const float* xyz = (const float*)d_in[0];
  int di = 1;
  auto nxt = [&]() { return (const float*)d_in[di++]; };

  const int   l1dims[3][3] = {{32, 32, 64}, {64, 64, 128}, {64, 96, 128}};
  const int   l1ns[3] = {16, 32, 128};
  const float l1r[3] = {0.1f, 0.2f, 0.4f};
  const int   l2dims[3][3] = {{64, 64, 128}, {128, 128, 256}, {128, 128, 256}};
  const int   l2ns[3] = {32, 64, 128};
  const float l2r[3] = {0.2f, 0.4f, 0.8f};
  const int   l3dims[4] = {256, 512, 1024, 128};

  const float *w_l1[3][3], *bb_l1[3][3], *w_l2[3][3], *bb_l2[3][3];
  const float *w_l3[4], *bb_l3[4];
  for (int i = 0; i < 3; i++)
    for (int j = 0; j < 3; j++) { w_l1[i][j] = nxt(); bb_l1[i][j] = nxt(); }
  for (int i = 0; i < 3; i++)
    for (int j = 0; j < 3; j++) { w_l2[i][j] = nxt(); bb_l2[i][j] = nxt(); }
  for (int j = 0; j < 4; j++) { w_l3[j] = nxt(); bb_l3[j] = nxt(); }

  char* ws = (char*)d_ws;
  size_t off = 0;
  auto alloc = [&](size_t bytes) -> void* {
    void* p = ws + off;
    off = (off + bytes + 255) & ~(size_t)255;
    return p;
  };

  int*   fpsIdx1 = (int*)alloc((size_t)BATCH * 512 * 4);
  float* newXyz1 = (float*)alloc((size_t)BATCH * 512 * 3 * 4);
  int*   idx1[3];
  for (int i = 0; i < 3; i++)
    idx1[i] = (int*)alloc((size_t)BATCH * 512 * l1ns[i] * 4);
  float* pts1 = (float*)alloc((size_t)BATCH * 512 * 320 * 4);
  int*   fpsIdx2 = (int*)alloc((size_t)BATCH * 128 * 4);
  float* newXyz2 = (float*)alloc((size_t)BATCH * 128 * 3 * 4);
  int*   idx2[3];
  for (int i = 0; i < 3; i++)
    idx2[i] = (int*)alloc((size_t)BATCH * 128 * l2ns[i] * 4);
  float* pts2 = (float*)alloc((size_t)BATCH * 128 * 640 * 4);

  _Float16 *wt_l1[3][3], *wt_l2[3][3], *wt_l3[4];
  float    *bp_l1[3][3], *bp_l2[3][3], *bp_l3[4];

  // ---- weight prep (f16 transpose + pad) ----
  for (int i = 0; i < 3; i++) {
    int K = 3;
    for (int j = 0; j < 3; j++) {
      int N = l1dims[i][j], Kp = rup32(K), Np = rup32(N);
      wt_l1[i][j] = (_Float16*)alloc((size_t)Kp * Np * 2);
      bp_l1[i][j] = (float*)alloc((size_t)Np * 4);
      int tot = Kp * Np;
      prep_weight_kernel<<<(tot + 255) / 256, 256, 0, stream>>>(
          w_l1[i][j], bb_l1[i][j], K, N, Kp, Np, wt_l1[i][j], bp_l1[i][j]);
      K = N;
    }
  }
  for (int i = 0; i < 3; i++) {
    int K = 323;   // 320 feats + 3 xyz
    for (int j = 0; j < 3; j++) {
      int N = l2dims[i][j], Kp = rup32(K), Np = rup32(N);
      wt_l2[i][j] = (_Float16*)alloc((size_t)Kp * Np * 2);
      bp_l2[i][j] = (float*)alloc((size_t)Np * 4);
      int tot = Kp * Np;
      prep_weight_kernel<<<(tot + 255) / 256, 256, 0, stream>>>(
          w_l2[i][j], bb_l2[i][j], K, N, Kp, Np, wt_l2[i][j], bp_l2[i][j]);
      K = N;
    }
  }
  {
    int K = 643;   // 3 xyz + 640 feats
    for (int j = 0; j < 4; j++) {
      int N = l3dims[j], Kp = rup32(K), Np = rup32(N);
      wt_l3[j] = (_Float16*)alloc((size_t)Kp * Np * 2);
      bp_l3[j] = (float*)alloc((size_t)Np * 4);
      int tot = Kp * Np;
      prep_weight_kernel<<<(tot + 255) / 256, 256, 0, stream>>>(
          w_l3[j], bb_l3[j], K, N, Kp, Np, wt_l3[j], bp_l3[j]);
      K = N;
    }
  }

  // ---- Stage 1: FPS(2048 -> 512), ball query, fused MSG ----
  fps_kernel<<<BATCH, 256, 0, stream>>>(xyz, NPTS0, 512, fpsIdx1, newXyz1);
  for (int i = 0; i < 3; i++)
    ball_query_kernel<<<(BATCH * 512 + 255) / 256, 256, 0, stream>>>(
        xyz, newXyz1, NPTS0, 512, l1r[i] * l1r[i], l1ns[i], idx1[i]);
  {
    int coff = 0;
    for (int i = 0; i < 3; i++) {
      fused_sa_kernel<96, 128, 32><<<BATCH * 512, 256, 0, stream>>>(
          xyz, nullptr, newXyz1, idx1[i], NPTS0, 512, l1ns[i], 0, 1,
          wt_l1[i][0], bp_l1[i][0], l1dims[i][0],
          wt_l1[i][1], bp_l1[i][1], l1dims[i][1],
          wt_l1[i][2], bp_l1[i][2], l1dims[i][2],
          nullptr, nullptr, 0, pts1, 320, coff);
      coff += l1dims[i][2];
    }
  }

  // ---- Stage 2: FPS(512 -> 128), ball query, fused MSG ----
  fps_kernel<<<BATCH, 256, 0, stream>>>(newXyz1, 512, 128, fpsIdx2, newXyz2);
  for (int i = 0; i < 3; i++)
    ball_query_kernel<<<(BATCH * 128 + 255) / 256, 256, 0, stream>>>(
        newXyz1, newXyz2, 512, 128, l2r[i] * l2r[i], l2ns[i], idx2[i]);
  {
    int coff = 0;
    for (int i = 0; i < 3; i++) {
      fused_sa_kernel<352, 256, 32><<<BATCH * 128, 256, 0, stream>>>(
          newXyz1, pts1, newXyz2, idx2[i], 512, 128, l2ns[i], 320, 1,
          wt_l2[i][0], bp_l2[i][0], l2dims[i][0],
          wt_l2[i][1], bp_l2[i][1], l2dims[i][1],
          wt_l2[i][2], bp_l2[i][2], l2dims[i][2],
          nullptr, nullptr, 0, pts2, 640, coff);
      coff += l2dims[i][2];
    }
  }

  // ---- Stage 3: group-all MLP [256,512,1024,128] + max -> d_out ----
  fused_sa_kernel<672, 1024, 16><<<BATCH, 256, 0, stream>>>(
      newXyz2, pts2, nullptr, nullptr, 128, 1, 128, 640, 0,
      wt_l3[0], bp_l3[0], l3dims[0],
      wt_l3[1], bp_l3[1], l3dims[1],
      wt_l3[2], bp_l3[2], l3dims[2],
      wt_l3[3], bp_l3[3], l3dims[3],
      (float*)d_out, 128, 0);

  // ---- L2 normalize embedding ----
  l2norm_kernel<<<BATCH, 128, 0, stream>>>((float*)d_out, 128);
}
